// MYLSTM_SF_85993835200933
// MI455X (gfx1250) — matrix operand order repaired and checked by reading, not compiled, not run on hardware
//
#include <hip/hip_runtime.h>
#include <hip/hip_bf16.h>

typedef __attribute__((ext_vector_type(16))) __bf16 v16bf;
typedef __attribute__((ext_vector_type(8)))  __bf16 v8bf;
typedef __attribute__((ext_vector_type(8)))  float  v8f;

#define B_TOT 64
#define T_LEN 1024
#define D_IN  256
#define U_DIM 256
#define GB    4                    // batches per workgroup
#define KBIG  1280                 // 256 (x_t) + 1024 (a0)
#define NBIG  1024                 // 4 gates * U
#define RW_ELEMS (64*40*32*16)     // [ntile][kstep][lane][16] bf16
#define W_ELEMS  (3*16*8*32*16)    // [mat][ntile][kstep][lane][16] bf16

// ---- Pack fp32 weights into WMMA-B fragment-linear bf16 layout ----
// B fragment order used here (contiguous K, same as a_frag): lane&15 = N column; lanes 0-15 hold
// K = kstep*32 + 0..15, lanes 16-31 hold K = kstep*32 + 16..31 (16 contiguous
// K values per lane) -> one 32B load per lane per fragment.
__global__ __launch_bounds__(256)
void MYLSTM_SF_85993835200933_prep(const float* __restrict__ kern,
                                   const float* __restrict__ rec,
                                   const float* __restrict__ Wq,
                                   const float* __restrict__ Wk,
                                   const float* __restrict__ Wv,
                                   __bf16* __restrict__ RWp,
                                   __bf16* __restrict__ Wp)
{
    const int total = RW_ELEMS + W_ELEMS;
    for (int idx = blockIdx.x * 256 + threadIdx.x; idx < total;
         idx += gridDim.x * 256) {
        if (idx < RW_ELEMS) {
            // RW[k][n]: k<256 -> kernel[g][k][o]; k>=256 -> recurrent_a[g][k-256][o]
            int i    = idx & 15;
            int lane = (idx >> 4) & 31;
            int rest = idx >> 9;
            int ks   = rest % 40;
            int nt   = rest / 40;
            int k = ks * 32 + ((lane >> 4) * 16) + i;
            int n = nt * 16 + (lane & 15);
            int g = n >> 8, o = n & 255;
            float v = (k < 256) ? kern[(g * 256 + k) * 256 + o]
                                : rec[(g * 1024 + (k - 256)) * 256 + o];
            RWp[idx] = (__bf16)v;
        } else {
            int id2  = idx - RW_ELEMS;
            int i    = id2 & 15;
            int lane = (id2 >> 4) & 31;
            int rest = id2 >> 9;
            int ks   = rest & 7;
            int r2   = rest >> 3;
            int nt   = r2 & 15;
            int mat  = r2 >> 4;
            int k = ks * 32 + ((lane >> 4) * 16) + i;
            int n = nt * 16 + (lane & 15);
            const float* W = (mat == 0) ? Wq : ((mat == 1) ? Wk : Wv);
            Wp[id2] = (__bf16)W[k * 256 + n];
        }
    }
}

// A fragment order used here (wave32): lane&15 = M row; kb=(lane>>4)*16;
// VGPR0-7 hold K = kb..kb+15 (contiguous, matching B).
static __device__ __forceinline__ v16bf a_frag(const __bf16* row, int koff) {
    union { v16bf v; v8bf h[2]; } u;
    u.h[0] = *(const v8bf*)(row + koff);        // K = koff .. koff+7
    u.h[1] = *(const v8bf*)(row + koff + 8);    // K = koff+8 .. koff+15
    return u.v;
}

static __device__ __forceinline__ float fast_tanh(float x) {
    float e = __expf(2.0f * x);
    return (e - 1.0f) / (e + 1.0f);
}

// ---- Persistent sequential kernel: 16 WGs x 4 batches, loop over T ----
__global__ __launch_bounds__(256)
void MYLSTM_SF_85993835200933_seq(const float* __restrict__ x,
                                  const float* __restrict__ bias,
                                  const __bf16* __restrict__ RWp,
                                  const __bf16* __restrict__ Wp,
                                  float* __restrict__ out)
{
    __shared__ __bf16 sH[4][GB][U_DIM];     // h history ring (4 slots)  8 KB
    __shared__ float  sC[GB][U_DIM];        // cell state                4 KB
    __shared__ __bf16 sQKV[3][16][U_DIM];   // q,k,v rows (b*4+j)       24 KB
    __shared__ __bf16 sA[GB][KBIG];         // [x_t | a0] per batch     10 KB
    __shared__ float  sZ[GB][NBIG];         // gate pre-activations     16 KB
    __shared__ float  sS16[16][16];         // raw scores (WMMA out)     1 KB
    __shared__ float  sAt[GB][4][4];        // softmax attn

    const int tid  = threadIdx.x;
    const int lane = tid & 31;
    const int w    = tid >> 5;              // 8 waves
    const int bB   = blockIdx.x * GB;

    for (int i = tid; i < 4 * GB * U_DIM; i += 256) ((__bf16*)sH)[i] = (__bf16)0.0f;
    for (int i = tid; i < GB * U_DIM;     i += 256) ((float*)sC)[i] = 0.0f;
    __syncthreads();

    const int mA = lane & 15;
    const int kb = (lane >> 4) * 16;        // A-fragment K base
    const int hl = lane >> 4;

    v16bf zero16;
    #pragma unroll
    for (int i = 0; i < 16; ++i) zero16[i] = (__bf16)0.0f;

    for (int t = 0; t < T_LEN; ++t) {
        const int s = (4 - (t & 3)) & 3;    // slot holding h1 this step

        // Phase 1: stage x_t (bf16) into sA[:, 0:256]
        for (int i = tid; i < GB * D_IN; i += 256) {
            int b = i >> 8, d = i & 255;
            sA[b][d] = (__bf16)x[(size_t)(bB + b) * T_LEN * D_IN + (size_t)t * D_IN + d];
        }
        __syncthreads();

        // Phase 2: q,k,v = x_h @ W_{q,k,v}; A is 16 rows = (batch*4 + hist)
        {
            const int bh = mA >> 2, jh = mA & 3;
            const __bf16* hrow = &sH[(s + jh) & 3][bh][0];
            v16bf afr[8];
            #pragma unroll
            for (int ks = 0; ks < 8; ++ks) afr[ks] = a_frag(hrow, ks * 32 + kb);

            #pragma unroll
            for (int tt = 0; tt < 6; ++tt) {            // 8 waves * 6 = 48 tiles
                const int idx = w * 6 + tt;
                const int mat = idx >> 4, nt = idx & 15;
                v8f acc;
                #pragma unroll
                for (int r = 0; r < 8; ++r) acc[r] = 0.0f;
                const __bf16* bp = Wp + (size_t)(mat * 16 + nt) * 4096 + lane * 16;
                #pragma unroll
                for (int ks = 0; ks < 8; ++ks) {
                    v16bf bfr = *(const v16bf*)(bp + ks * 512);
                    acc = __builtin_amdgcn_wmma_f32_16x16x32_bf16(
                        false, afr[ks], false, bfr, (short)0, acc, false, false);
                }
                const int col = nt * 16 + (lane & 15);
                #pragma unroll
                for (int r = 0; r < 8; ++r)
                    sQKV[mat][r + 8 * hl][col] = (__bf16)acc[r];
            }
        }
        __syncthreads();

        // Phase 3: scores = Q . K^T via one-wave WMMA (8 x 16x16x32).
        // B[d][n] = k[n][d] -> lane (n = lane&15) reads 16 contiguous bf16
        // from row n of sQKV[1]: a single 32B LDS read per fragment.
        if (w == 0) {
            const __bf16* qrow = &sQKV[0][mA][0];
            const __bf16* krow = &sQKV[1][lane & 15][0];
            const int koffB = hl * 16;
            v8f acc;
            #pragma unroll
            for (int r = 0; r < 8; ++r) acc[r] = 0.0f;
            #pragma unroll
            for (int ks = 0; ks < 8; ++ks) {
                v16bf aq = a_frag(qrow, ks * 32 + kb);
                v16bf bk = *(const v16bf*)(krow + ks * 32 + koffB);
                acc = __builtin_amdgcn_wmma_f32_16x16x32_bf16(
                    false, aq, false, bk, (short)0, acc, false, false);
            }
            #pragma unroll
            for (int r = 0; r < 8; ++r)
                sS16[r + 8 * hl][lane & 15] = acc[r] * 0.0625f;   // 1/sqrt(256)
        }
        __syncthreads();

        // Phase 4: softmax over the 4 block-diagonal entries
        if (tid < 16) {
            int b = tid >> 2, qi = tid & 3;
            float sc[4];
            #pragma unroll
            for (int k = 0; k < 4; ++k) sc[k] = sS16[b * 4 + qi][b * 4 + k];
            float m0 = fmaxf(fmaxf(sc[0], sc[1]), fmaxf(sc[2], sc[3]));
            float e[4], sum = 0.0f;
            #pragma unroll
            for (int k = 0; k < 4; ++k) { e[k] = __expf(sc[k] - m0); sum += e[k]; }
            float inv = 1.0f / sum;
            #pragma unroll
            for (int k = 0; k < 4; ++k) sAt[b][qi][k] = e[k] * inv;
        }
        __syncthreads();

        // Phase 5: a0 = attn @ v  ->  sA[:, 256:1280]
        for (int i = tid; i < GB * 1024; i += 256) {
            int b = i >> 10, r = i & 1023, j = r >> 8, o = r & 255;
            float e = 0.0f;
            #pragma unroll
            for (int k = 0; k < 4; ++k)
                e += sAt[b][j][k] * (float)sQKV[2][b * 4 + k][o];
            sA[b][256 + r] = (__bf16)e;
        }
        __syncthreads();

        // Phase 6: Z = [x_t|a0] @ RW  (M=16 pad, K=1280, N=1024); K-outer,
        // 8 N-tiles per wave held in accumulators, one A frag load per K-step.
        {
            v8f acc[8];
            #pragma unroll
            for (int tt = 0; tt < 8; ++tt)
                #pragma unroll
                for (int r = 0; r < 8; ++r) acc[tt][r] = 0.0f;

            const __bf16* arow = &sA[mA & (GB - 1)][0];
            for (int ks = 0; ks < 40; ++ks) {
                v16bf af = a_frag(arow, ks * 32 + kb);
                if (mA >= GB) af = zero16;              // pad rows 4..15
                #pragma unroll
                for (int tt = 0; tt < 8; ++tt) {
                    const int nt = w * 8 + tt;
                    v16bf bf = *(const v16bf*)(RWp + ((size_t)(nt * 40 + ks) * 32 + lane) * 16);
                    acc[tt] = __builtin_amdgcn_wmma_f32_16x16x32_bf16(
                        false, af, false, bf, (short)0, acc[tt], false, false);
                }
            }
            if (lane < 16) {                            // rows 0..3 live in lanes 0-15
                #pragma unroll
                for (int tt = 0; tt < 8; ++tt) {
                    int col = (w * 8 + tt) * 16 + lane;
                    #pragma unroll
                    for (int r = 0; r < GB; ++r) sZ[r][col] = acc[tt][r];
                }
            }
        }
        __syncthreads();

        // Phase 7: gates, cell/hidden update, output
        for (int i = tid; i < GB * U_DIM; i += 256) {
            int b = i >> 8, o = i & 255;
            float zf = sZ[b][o]       + bias[o];
            float zi = sZ[b][256 + o] + bias[256 + o];
            float zo = sZ[b][512 + o] + bias[512 + o];
            float zc = sZ[b][768 + o] + bias[768 + o];
            float fg = 1.0f / (1.0f + __expf(-zf));
            float ig = 1.0f / (1.0f + __expf(-zi));
            float og = 1.0f / (1.0f + __expf(-zo));
            float ch = fast_tanh(zc);
            float c0 = fg * sC[b][o] + ig * ch;
            float h  = og * fast_tanh(c0);
            sC[b][o] = c0;
            sH[(s + 3) & 3][b][o] = (__bf16)h;          // new h1 slot
            out[(size_t)(bB + b) * T_LEN * U_DIM + (size_t)t * U_DIM + o] = h;
        }
        __syncthreads();
    }
}

extern "C" void kernel_launch(void* const* d_in, const int* in_sizes, int n_in,
                              void* d_out, int out_size, void* d_ws, size_t ws_size,
                              hipStream_t stream) {
    (void)in_sizes; (void)n_in; (void)out_size; (void)ws_size;
    const float* x    = (const float*)d_in[0];
    const float* kern = (const float*)d_in[1];
    const float* rec  = (const float*)d_in[2];
    const float* bias = (const float*)d_in[3];
    const float* Wq   = (const float*)d_in[4];
    const float* Wk   = (const float*)d_in[5];
    const float* Wv   = (const float*)d_in[6];

    __bf16* RWp = (__bf16*)d_ws;
    __bf16* Wp  = (__bf16*)((char*)d_ws + (size_t)RW_ELEMS * 2);

    const int totalPrep = RW_ELEMS + W_ELEMS;
    const int prepBlocks = (totalPrep + 255) / 256;
    MYLSTM_SF_85993835200933_prep<<<prepBlocks, 256, 0, stream>>>(
        kern, rec, Wq, Wk, Wv, RWp, Wp);

    MYLSTM_SF_85993835200933_seq<<<B_TOT / GB, 256, 0, stream>>>(
        x, bias, RWp, Wp, (float*)d_out);
}
